// HyperGraph_77077483094346
// MI455X (gfx1250) — matrix operand order repaired and checked by reading, not compiled, not run on hardware
//
#include <hip/hip_runtime.h>

// ---------------------------------------------------------------------------
// HyperGNN on MI455X (gfx1250, wave32).
//  * H (binary) converted once to f16 (exact) + f16 transpose -> 200MB streams
//  * Big GEMMs via V_WMMA_F32_16X16X32_F16, 16x128 tile per 256-thr block,
//    K-step 64 (2 WMMA per staging phase), register-transposed B staging
//    with ds_store_b128 only, global_prefetch on streaming tiles.
//  * Node GEMM fuses dv_inv scale + emb add + LayerNorm epilogue (LDS reuse)
//  * Edge GEMM fuses de_inv / 1/max(cnt,1) scale
// Requires ws_size >= ~410 MB.
// ---------------------------------------------------------------------------

typedef __attribute__((ext_vector_type(16))) _Float16 v16h;
typedef __attribute__((ext_vector_type(8)))  _Float16 v8h;
typedef __attribute__((ext_vector_type(4)))  _Float16 v4h;
typedef __attribute__((ext_vector_type(8)))  float    v8f;

constexpr int   Nn      = 20000;
constexpr int   Ee      = 5000;
constexpr int   Dd      = 128;
constexpr int   ATT     = 32;
constexpr int   LAYERS  = 4;
constexpr float LN_EPS  = 1e-5f;
constexpr float DEG_EPS = 1e-12f;
constexpr int   KB      = 64;   // K tile (2 WMMA k-steps)
constexpr int   BS      = 72;   // padded col stride (elems) for B tile; 144B, 16B-aligned

// ---------------------------------------------------------------------------
// Core: one 16(rows) x 128(cols) f32 tile of A[MxK] @ B[KxD], K-loop step 64.
// Block = 256 threads = 8 waves; wave w owns cols [16w, 16w+16).
// ---------------------------------------------------------------------------
static __device__ __forceinline__
v8f gemm16x128_core(const _Float16* __restrict__ A, int lda, int row0, int rows,
                    const _Float16* __restrict__ B, int Kdim,
                    _Float16* shA, _Float16* shB)
{
  const int tid  = threadIdx.x;
  const int wave = tid >> 5;
  const int lane = tid & 31;
  const int lcol = lane & 15;
  const int hi   = lane >> 4;
  const int gcol = wave * 16 + lcol;

  // A staging (256 threads): row = t/16, k-chunk = (t%16)*4   -> 16x64 tile
  const int ar = tid >> 4;
  const int ak = (tid & 15) * 4;
  // B staging (256 threads): k-quad = (t/32)*8, cols = (t%32)*4 -> 64x128 tile
  const int bkq = (tid >> 5) * 8;
  const int bc  = (tid & 31) * 4;

  v8f c = {};

  for (int k0 = 0; k0 < Kdim; k0 += KB) {
    if (k0) __syncthreads();

    // ---- stage A tile: 16 x 64, row-major (stride 64)
    {
      const int grow = row0 + ar;
      v4h av = {};
      if (grow < rows) {
        const _Float16* src = A + (size_t)grow * lda + k0 + ak;
        if (k0 + ak + 3 < Kdim) {
          av = *(const v4h*)src;
          __builtin_prefetch(src + KB, 0, 1);        // next K tile of A
        } else {
          #pragma unroll
          for (int i = 0; i < 4; ++i)
            if (k0 + ak + i < Kdim) av[i] = src[i];
        }
      }
      *(v4h*)(shA + ar * KB + ak) = av;
    }

    // ---- stage B tile: 64 x 128 -> column-major LDS (stride BS),
    //      8(K) x 4(col) sub-block per thread, transposed in registers,
    //      written as 4x ds_store_b128.
    {
      v4h rj[8];
      #pragma unroll
      for (int j = 0; j < 8; ++j) {
        rj[j] = v4h{};
        if (k0 + bkq + j < Kdim)
          rj[j] = *(const v4h*)(B + (size_t)(k0 + bkq + j) * Dd + bc);
      }
      if (k0 + KB < Kdim)
        __builtin_prefetch(B + (size_t)(k0 + bkq + KB) * Dd + bc, 0, 1);
      #pragma unroll
      for (int i = 0; i < 4; ++i) {
        v8h col;
        #pragma unroll
        for (int j = 0; j < 8; ++j) col[j] = rj[j][i];
        *(v8h*)(shB + (bc + i) * BS + bkq) = col;    // 16B aligned
      }
    }
    __syncthreads();

    // ---- compute: 2 WMMA k-steps per staged tile
    #pragma unroll
    for (int s = 0; s < 2; ++s) {
      // A frag: lane row = lane%16; k = s*32 + (hi?16:0) + {0..15}
      const _Float16* ap = shA + lcol * KB + s * 32 + hi * 16;
      v8h aA = *(const v8h*)(ap);
      v8h aB = *(const v8h*)(ap + 8);
      // B frag: lane col = lane%16; k = s*32 + (hi?16:0) + {0..15}
      const _Float16* bp = shB + gcol * BS + s * 32 + hi * 16;
      v8h bA = *(const v8h*)(bp);
      v8h bB = *(const v8h*)(bp + 8);
      v16h a, b;
      #pragma unroll
      for (int i = 0; i < 8; ++i) {
        a[i] = aA[i]; a[i + 8] = aB[i];
        b[i] = bA[i]; b[i + 8] = bB[i];
      }
      c = __builtin_amdgcn_wmma_f32_16x16x32_f16(
              false, a, false, b, (short)0, c, false, false);
    }
  }
  __syncthreads();   // allow epilogue to reuse shB
  return c;
}

// ---------------------------------------------------------------------------
// Edge-side GEMM: out[e,:] = (HT @ xb)[e,:] * rowscale[e]
// ---------------------------------------------------------------------------
__global__ __launch_bounds__(256)
void k_gemm_edge(const _Float16* __restrict__ HT, const _Float16* __restrict__ xb,
                 const float* __restrict__ rowscale,
                 _Float16* __restrict__ outH, float* __restrict__ outF)
{
  __shared__ __align__(16) _Float16 shA[16 * KB];
  __shared__ __align__(16) _Float16 shB[Dd * BS];
  const int row0 = blockIdx.x * 16;
  v8f c = gemm16x128_core(HT, Nn, row0, Ee, xb, Nn, shA, shB);

  const int lane = threadIdx.x & 31;
  const int hi   = lane >> 4;
  const int lcol = lane & 15;
  const int gcol = (threadIdx.x >> 5) * 16 + lcol;
  #pragma unroll
  for (int r = 0; r < 8; ++r) {
    int gr = row0 + r + hi * 8;                  // C layout: row = r + 8*hi
    if (gr < Ee) {
      float v = c[r] * rowscale[gr];
      if (outH) outH[(size_t)gr * Dd + gcol] = (_Float16)v;
      if (outF) outF[(size_t)gr * Dd + gcol] = v;
    }
  }
}

// ---------------------------------------------------------------------------
// Node-side GEMM + fused bias + LayerNorm: xb_out = LN(c*dv_inv + emb)
// ---------------------------------------------------------------------------
__global__ __launch_bounds__(256)
void k_gemm_node(const _Float16* __restrict__ Hh, const _Float16* __restrict__ eb,
                 const float* __restrict__ dv_inv, const float* __restrict__ emb,
                 const float* __restrict__ gamma, const float* __restrict__ beta,
                 _Float16* __restrict__ xb_out)
{
  __shared__ __align__(16) _Float16 shA[16 * KB];
  __shared__ __align__(16) _Float16 shB[Dd * BS];
  __shared__ float shPS[16][17], shPS2[16][17], shStats[16][2];
  const int row0 = blockIdx.x * 16;              // Nn % 16 == 0, no guard
  v8f c = gemm16x128_core(Hh, Ee, row0, Nn, eb, Ee, shA, shB);

  float* shT = (float*)shB;                      // 16x128 f32 (8KB <= 18KB)
  const int tid  = threadIdx.x;
  const int lane = tid & 31;
  const int hi   = lane >> 4;
  const int lcol = lane & 15;
  const int gcol = (tid >> 5) * 16 + lcol;
  #pragma unroll
  for (int r = 0; r < 8; ++r) {
    int lr = r + hi * 8;
    int gr = row0 + lr;
    shT[lr * Dd + gcol] = c[r] * dv_inv[gr] + emb[(size_t)gr * Dd + gcol];
  }
  __syncthreads();

  const int row = tid >> 4, g = tid & 15;        // 16 rows x 16 groups
  float s = 0.f, s2 = 0.f;
  #pragma unroll
  for (int i = 0; i < 8; ++i) {
    float v = shT[row * Dd + g * 8 + i];
    s += v; s2 += v * v;
  }
  shPS[row][g] = s; shPS2[row][g] = s2;
  __syncthreads();
  if (g == 0) {
    float S = 0.f, S2 = 0.f;
    #pragma unroll
    for (int i = 0; i < 16; ++i) { S += shPS[row][i]; S2 += shPS2[row][i]; }
    float mu  = S * (1.0f / Dd);
    float var = S2 * (1.0f / Dd) - mu * mu;
    shStats[row][0] = mu;
    shStats[row][1] = rsqrtf(var + LN_EPS);
  }
  __syncthreads();
  float mu = shStats[row][0], rs = shStats[row][1];
  #pragma unroll
  for (int i = 0; i < 8; ++i) {
    int cix = g * 8 + i;
    float v = (shT[row * Dd + cix] - mu) * rs * gamma[cix] + beta[cix];
    xb_out[(size_t)(row0 + row) * Dd + cix] = (_Float16)v;
  }
}

// ---------------------------------------------------------------------------
// Degrees (edge degrees parallelized over N with float atomics)
// ---------------------------------------------------------------------------
__global__ void k_zero(float* __restrict__ p, int n)
{
  int i = blockIdx.x * blockDim.x + threadIdx.x;
  if (i < n) p[i] = 0.f;
}

__global__ void k_deg_edge_part(const float* __restrict__ H, float* __restrict__ de)
{
  int e = blockIdx.x * 256 + threadIdx.x;
  if (e >= Ee) return;
  const int n0 = blockIdx.y * 250;               // 80 chunks x 250 rows
  const float* p = H + (size_t)n0 * Ee + e;
  float s = 0.f;
  #pragma unroll 5
  for (int n = 0; n < 250; ++n) s += p[(size_t)n * Ee];
  atomicAdd(&de[e], s);
}

__global__ void k_deg_edge_fin(const float* __restrict__ de,
                               float* __restrict__ de_inv, float* __restrict__ cnt_inv)
{
  int e = blockIdx.x * blockDim.x + threadIdx.x;
  if (e >= Ee) return;
  float s = de[e];
  de_inv[e]  = 1.0f / fmaxf(s, DEG_EPS);
  cnt_inv[e] = 1.0f / fmaxf(s, 1.0f);
}

__global__ void k_deg_node(const float* __restrict__ H, float* __restrict__ dv_inv)
{
  int node = (blockIdx.x * blockDim.x + threadIdx.x) >> 5;
  int lane = threadIdx.x & 31;
  if (node >= Nn) return;
  const float* p = H + (size_t)node * Ee;
  float s = 0.f;
  for (int e = lane; e < Ee; e += 32) s += p[e];
  #pragma unroll
  for (int off = 16; off; off >>= 1) s += __shfl_xor(s, off, 32);
  if (lane == 0) dv_inv[node] = 1.0f / fmaxf(s, DEG_EPS);
}

// ---------------------------------------------------------------------------
// fp32 -> f16 (vectorized x4); used for Hh and initial xb
// ---------------------------------------------------------------------------
__global__ void k_f2h(const float* __restrict__ src, _Float16* __restrict__ dst, long n4)
{
  long i = (long)blockIdx.x * blockDim.x + threadIdx.x;
  if (i >= n4) return;
  float4 v = ((const float4*)src)[i];
  v4h h;
  h[0] = (_Float16)v.x; h[1] = (_Float16)v.y;
  h[2] = (_Float16)v.z; h[3] = (_Float16)v.w;
  ((v4h*)dst)[i] = h;
}

// ---------------------------------------------------------------------------
// H[N,E] fp32 -> HT[E,N] f16, 32x32 LDS tiles
// ---------------------------------------------------------------------------
__global__ __launch_bounds__(256)
void k_transpose(const float* __restrict__ H, _Float16* __restrict__ HT)
{
  __shared__ _Float16 t[32][33];
  int e0 = blockIdx.x * 32, n0 = blockIdx.y * 32;
  int tx = threadIdx.x, ty = threadIdx.y;        // 32 x 8
  #pragma unroll
  for (int j = 0; j < 4; ++j) {
    int n = n0 + ty + j * 8, e = e0 + tx;
    t[ty + j * 8][tx] = (e < Ee) ? (_Float16)H[(size_t)n * Ee + e] : (_Float16)0.f;
  }
  __syncthreads();
  #pragma unroll
  for (int j = 0; j < 4; ++j) {
    int e = e0 + ty + j * 8, n = n0 + tx;
    if (e < Ee) HT[(size_t)e * Nn + n] = t[tx][ty + j * 8];
  }
}

// ---------------------------------------------------------------------------
// Attention logits: z[e] = tanh(edge_out[e] @ W + b) . q
// ---------------------------------------------------------------------------
__global__ __launch_bounds__(128)
void k_att_logits(const float* __restrict__ EO, const float* __restrict__ W,
                  const float* __restrict__ bA, const float* __restrict__ q,
                  float* __restrict__ z)
{
  __shared__ float row[Dd];
  int e = blockIdx.x, t = threadIdx.x;
  row[t] = EO[(size_t)e * Dd + t];
  __syncthreads();
  if (t < ATT) {
    float s = bA[t];
    #pragma unroll 4
    for (int d = 0; d < Dd; ++d) s += row[d] * W[d * ATT + t];
    float v = tanhf(s) * q[t];
    #pragma unroll
    for (int off = 16; off; off >>= 1) v += __shfl_xor(v, off, 32);
    if (t == 0) z[e] = v;
  }
}

__global__ __launch_bounds__(256)
void k_sm_stats(const float* __restrict__ z, float* __restrict__ stats)
{
  __shared__ float red[256];
  int t = threadIdx.x;
  float m = -1e30f;
  for (int e = t; e < Ee; e += 256) m = fmaxf(m, z[e]);
  red[t] = m; __syncthreads();
  for (int s = 128; s; s >>= 1) { if (t < s) red[t] = fmaxf(red[t], red[t + s]); __syncthreads(); }
  float zmax = red[0]; __syncthreads();
  float acc = 0.f;
  for (int e = t; e < Ee; e += 256) acc += expf(z[e] - zmax);
  red[t] = acc; __syncthreads();
  for (int s = 128; s; s >>= 1) { if (t < s) red[t] += red[t + s]; __syncthreads(); }
  if (t == 0) { stats[0] = zmax; stats[1] = red[0]; }
}

__global__ __launch_bounds__(128)
void k_pool(const float* __restrict__ EO, const float* __restrict__ z,
            const float* __restrict__ stats, float* __restrict__ out)
{
  int d = threadIdx.x;
  int e0 = blockIdx.x * 125;
  int e1 = (e0 + 125 < Ee) ? e0 + 125 : Ee;
  float zmax = stats[0], inv = 1.0f / stats[1];
  float acc = 0.f;
  for (int e = e0; e < e1; ++e)
    acc += expf(z[e] - zmax) * EO[(size_t)e * Dd + d];
  atomicAdd(&out[d], acc * inv);
}

// ---------------------------------------------------------------------------
extern "C" void kernel_launch(void* const* d_in, const int* in_sizes, int n_in,
                              void* d_out, int out_size, void* d_ws, size_t ws_size,
                              hipStream_t stream)
{
  (void)in_sizes; (void)n_in; (void)out_size; (void)ws_size;
  const float* emb  = (const float*)d_in[0];
  const float* H    = (const float*)d_in[1];
  const float* gam  = (const float*)d_in[2];
  const float* bet  = (const float*)d_in[3];
  const float* Watt = (const float*)d_in[4];
  const float* batt = (const float*)d_in[5];
  const float* qatt = (const float*)d_in[6];
  float* out = (float*)d_out;

  char* w = (char*)d_ws;
  auto carve = [&](size_t bytes) {
    char* p = w; w += (bytes + 255) & ~(size_t)255; return p;
  };
  const size_t NE = (size_t)Nn * Ee;
  _Float16* Hh   = (_Float16*)carve(NE * 2);               // 200 MB
  _Float16* HTh  = (_Float16*)carve(NE * 2);               // 200 MB
  _Float16* xb   = (_Float16*)carve((size_t)Nn * Dd * 2);  // 5.12 MB
  _Float16* eb   = (_Float16*)carve((size_t)Ee * Dd * 2);  // 1.28 MB
  float* EO      = (float*)carve((size_t)Ee * Dd * 4);     // 2.56 MB
  float* de_acc  = (float*)carve((size_t)Ee * 4);
  float* de_inv  = (float*)carve((size_t)Ee * 4);
  float* cnt_inv = (float*)carve((size_t)Ee * 4);
  float* dv_inv  = (float*)carve((size_t)Nn * 4);
  float* z       = (float*)carve((size_t)Ee * 4);
  float* stats   = (float*)carve(2 * 4);

  // ---- setup: degrees, f16 copies of H / H^T, initial x (f16 of emb)
  k_zero<<<1, 128, 0, stream>>>(out, Dd);
  k_zero<<<(Ee + 255) / 256, 256, 0, stream>>>(de_acc, Ee);
  {
    dim3 dg((Ee + 255) / 256, 80);                         // 80 x 250 rows
    k_deg_edge_part<<<dg, 256, 0, stream>>>(H, de_acc);
  }
  k_deg_edge_fin<<<(Ee + 255) / 256, 256, 0, stream>>>(de_acc, de_inv, cnt_inv);
  k_deg_node<<<Nn / 8, 256, 0, stream>>>(H, dv_inv);
  {
    long n4 = (long)(NE / 4);
    k_f2h<<<(int)((n4 + 255) / 256), 256, 0, stream>>>(H, Hh, n4);
  }
  {
    dim3 tg((Ee + 31) / 32, Nn / 32);
    k_transpose<<<tg, dim3(32, 8), 0, stream>>>(H, HTh);
  }
  {
    long n4 = (long)Nn * Dd / 4;
    k_f2h<<<(int)((n4 + 255) / 256), 256, 0, stream>>>(emb, xb, n4);
  }

  // ---- 4 layers of V->E mean, E->V mean, fused bias+LayerNorm
  const int egrid = (Ee + 15) / 16;   // 313
  const int ngrid = Nn / 16;          // 1250
  for (int l = 0; l < LAYERS; ++l) {
    k_gemm_edge<<<egrid, 256, 0, stream>>>(HTh, xb, de_inv, eb, nullptr);
    k_gemm_node<<<ngrid, 256, 0, stream>>>(Hh, eb, dv_inv, emb, gam, bet, xb);
  }

  // ---- final per-edge masked mean (scale = 1/max(cnt,1)), f32 output
  k_gemm_edge<<<egrid, 256, 0, stream>>>(HTh, xb, cnt_inv, nullptr, EO);

  // ---- attention pooling
  k_att_logits<<<Ee, 128, 0, stream>>>(EO, Watt, batt, qatt, z);
  k_sm_stats<<<1, 256, 0, stream>>>(z, stats);
  k_pool<<<40, 128, 0, stream>>>(EO, z, stats, out);
}
